// Attention_17411797418011
// MI455X (gfx1250) — compile-verified, matrix operand order checked by reading, not run on hardware
//
#include <hip/hip_runtime.h>

typedef __attribute__((ext_vector_type(4)))  __bf16 v4bf;
typedef __attribute__((ext_vector_type(8)))  __bf16 v8bf;
typedef __attribute__((ext_vector_type(16))) __bf16 v16bf;
typedef __attribute__((ext_vector_type(8)))  float  v8f;

__device__ inline v16bf bf_combine(v8bf lo, v8bf hi) {
  return __builtin_shufflevector(lo, hi, 0,1,2,3,4,5,6,7,8,9,10,11,12,13,14,15);
}
__device__ inline v8bf ld8(const __bf16* p) { return *(const v8bf*)p; }
__device__ inline v8bf cvt8(const float* p) {
  const float4* q = (const float4*)p;
  float4 a = q[0], b = q[1];
  v8bf r;
  r[0]=(__bf16)a.x; r[1]=(__bf16)a.y; r[2]=(__bf16)a.z; r[3]=(__bf16)a.w;
  r[4]=(__bf16)b.x; r[5]=(__bf16)b.y; r[6]=(__bf16)b.z; r[7]=(__bf16)b.w;
  return r;
}
__device__ inline v8f wmma_bf16(v16bf a, v16bf b, v8f c) {
  return __builtin_amdgcn_wmma_f32_16x16x32_bf16(false, a, false, b, (short)0, c, false, false);
}

// ---------------- prep: fp32 -> bf16 weight/q conversion (+ Wk transpose) ---
__global__ __launch_bounds__(256)
void prep_kernel(const float* __restrict__ q,  const float* __restrict__ Wq,
                 const float* __restrict__ Wk, const float* __restrict__ Wv,
                 const float* __restrict__ Wb, const float* __restrict__ Wo,
                 __bf16* qbf, __bf16* Wqb, __bf16* WkT,
                 __bf16* Wvb, __bf16* Wbb, __bf16* Wob)
{
  int i = blockIdx.x * 256 + threadIdx.x;      // grid covers 2M
  if (i < 2048*1024) qbf[i] = (__bf16)q[i];
  if (i < 1024*1024) {
    Wqb[i] = (__bf16)Wq[i];
    Wvb[i] = (__bf16)Wv[i];
    Wob[i] = (__bf16)Wo[i];
    int r = i >> 10, c = i & 1023;
    WkT[(long)c*1024 + r] = (__bf16)Wk[i];      // WkT[c][r] = Wk[r][c]
  }
  if (i < 128*128) Wbb[i] = (__bf16)Wb[i];
}

// ---------------- generic  C(16xN per block) = A(MxK) @ Bt(NxK)^T -----------
// blockIdx.z selects a "head" via pointer strides zA/zB/zC.
__global__ __launch_bounds__(128)
void gemm16_kernel(const __bf16* __restrict__ A, int lda, long zA,
                   const __bf16* __restrict__ Bt, int ldb, long zB,
                   __bf16* __restrict__ Cbf, float* __restrict__ Cf,
                   const float* __restrict__ bias,
                   int ldc, long zC, int K)
{
  __shared__ __attribute__((aligned(16))) __bf16 As[16 * 1024];
  const int tid = threadIdx.x, lane = tid & 31, wave = tid >> 5;
  const int m0 = blockIdx.y * 16;
  const int n0 = blockIdx.x * 64 + wave * 16;
  const int z  = blockIdx.z;

  A  += (long)z * zA + (long)m0 * lda;
  Bt += (long)z * zB + (long)n0 * ldb;

  for (int i = tid * 8; i < 16 * K; i += 128 * 8) {   // stage 16 x K A-tile
    int r = i / K, c = i % K;
    *(v8bf*)(As + r * K + c) = *(const v8bf*)(A + (long)r * lda + c);
  }
  __syncthreads();

  v8f acc = {};
  const __bf16* aRow = As + (lane & 15) * K;
  const __bf16* bRow = Bt + (long)(lane & 15) * ldb;
  const int sub = (lane >> 4) * 8;
  for (int k0 = 0; k0 < K; k0 += 32) {
    const __bf16* ap = aRow + k0 + sub;
    const __bf16* bp = bRow + k0 + sub;
    v16bf af = bf_combine(ld8(ap), ld8(ap + 16));
    v16bf bf = bf_combine(ld8(bp), ld8(bp + 16));
    acc = wmma_bf16(af, bf, acc);
  }

  const int col = n0 + (lane & 15);
  const int rb  = m0 + ((lane >> 4) ? 8 : 0);
  const long cb = (long)z * zC;
  if (Cf) {
    float bv = bias ? bias[col] : 0.f;
    #pragma unroll
    for (int r = 0; r < 8; r++) Cf[cb + (long)(rb + r) * ldc + col] = acc[r] + bv;
  } else {
    #pragma unroll
    for (int r = 0; r < 8; r++) Cbf[cb + (long)(rb + r) * ldc + col] = (__bf16)acc[r];
  }
}

// ---------------- fused streaming attention: scores + softmax + vbar --------
// One workgroup per batch b: streams K_b, V_b (fp32, 512 KB) exactly once.
#define VT_COLS   128
#define VT_STRIDE 72     // halves; 144B rows -> 8B aligned, spread over banks

// stage a 64 x 128 fp32 tile of V (columns [p*128, p*128+128)) into LDS as
// bf16, transposed; 4x4 micro-tiles: 4x global_load_b128 + 4x ds_store_b64
__device__ inline void stage_v(const float* __restrict__ Vb, __bf16* dst,
                               int p, int tstart, int nt)
{
  for (int t = tstart; t < 512; t += nt) {         // 16 mtiles x 32 ctiles
    int m0 = (t >> 5) << 2, c0 = (t & 31) << 2;
    const float* s = Vb + (long)m0 * 1024 + p * VT_COLS + c0;
    float4 r0 = *(const float4*)(s);
    float4 r1 = *(const float4*)(s + 1024);
    float4 r2 = *(const float4*)(s + 2048);
    float4 r3 = *(const float4*)(s + 3072);
    __bf16* d = dst + c0 * VT_STRIDE + m0;
    v4bf w;
    w[0]=(__bf16)r0.x; w[1]=(__bf16)r1.x; w[2]=(__bf16)r2.x; w[3]=(__bf16)r3.x;
    *(v4bf*)(d) = w;
    w[0]=(__bf16)r0.y; w[1]=(__bf16)r1.y; w[2]=(__bf16)r2.y; w[3]=(__bf16)r3.y;
    *(v4bf*)(d + VT_STRIDE) = w;
    w[0]=(__bf16)r0.z; w[1]=(__bf16)r1.z; w[2]=(__bf16)r2.z; w[3]=(__bf16)r3.z;
    *(v4bf*)(d + 2*VT_STRIDE) = w;
    w[0]=(__bf16)r0.w; w[1]=(__bf16)r1.w; w[2]=(__bf16)r2.w; w[3]=(__bf16)r3.w;
    *(v4bf*)(d + 3*VT_STRIDE) = w;
  }
}

__global__ __launch_bounds__(256)
void attn_kernel(const float* __restrict__ Kg, const float* __restrict__ Vg,
                 const __bf16* __restrict__ T, __bf16* __restrict__ VB)
{
  __shared__ __attribute__((aligned(16))) __bf16 Vt0[VT_COLS * VT_STRIDE]; // 18 KB
  __shared__ __attribute__((aligned(16))) __bf16 Vt1[VT_COLS * VT_STRIDE]; // 18 KB
  __shared__ float  sc[64 * 16];                                           // 4 KB
  __shared__ __attribute__((aligned(16))) __bf16 attnA[16 * 64];           // 2 KB

  const int tid = threadIdx.x, lane = tid & 31, wave = tid >> 5;
  const int b = blockIdx.x;
  const int sub = (lane >> 4) * 8;
  const float*  Kb = Kg + (long)b * 64 * 1024;
  const float*  Vb = Vg + (long)b * 64 * 1024;
  const __bf16* Tb = T  + (long)b * 8 * 1024;

  // ---- scores[m,h] = sum_c K_b[m,c] * T_b[h,c]   (waves 0..3, m-tile each)
  //      meanwhile waves 4..7 stage V pass 0 -> overlap K-stream and V-stream
  if (wave < 4) {
    v8f acc = {};
    const float*  aRow = Kb + (long)(wave * 16 + (lane & 15)) * 1024;
    const __bf16* bRow = Tb + (lane & 7) * 1024;   // cols 8..15 mirror heads
    for (int k0 = 0; k0 < 1024; k0 += 32) {
      if (k0 < 768) __builtin_prefetch(aRow + k0 + 256, 0, 1); // global_prefetch
      int off = k0 + sub;
      v16bf af = bf_combine(cvt8(aRow + off), cvt8(aRow + off + 16));
      v16bf bf = bf_combine(ld8(bRow + off), ld8(bRow + off + 16));
      acc = wmma_bf16(af, bf, acc);
    }
    int col = lane & 15, rb = wave * 16 + ((lane >> 4) ? 8 : 0);
    #pragma unroll
    for (int r = 0; r < 8; r++) sc[(rb + r) * 16 + col] = acc[r];
  } else {
    stage_v(Vb, Vt0, 0, tid - 128, 128);
  }
  __syncthreads();

  // ---- softmax over m per head (8 threads; tids 8..15 zero pad rows); waves 4..7 stage V pass 1
  if (tid < 8) {
    float mx = -1e30f;
    for (int m = 0; m < 64; m++) mx = fmaxf(mx, sc[m * 16 + tid]);
    float s = 0.f;
    for (int m = 0; m < 64; m++) s += __expf(sc[m * 16 + tid] - mx);
    float inv = 1.f / s;
    for (int m = 0; m < 64; m++)
      attnA[tid * 64 + m] = (__bf16)(__expf(sc[m * 16 + tid] - mx) * inv);
  } else if (tid < 16) {
    for (int m = 0; m < 64; m++) attnA[tid * 64 + m] = (__bf16)0.f;
  } else if (wave >= 4) {
    stage_v(Vb, Vt1, 1, tid - 128, 128);
  }
  __syncthreads();

  // ---- vbar[h,c] = sum_m attn[h,m] * V_b[m,c]; double-buffered column passes
  const __bf16* aRow = attnA + (lane & 15) * 64;
  for (int p = 0; p < 1024 / VT_COLS; p++) {
    const __bf16* buf = (p & 1) ? Vt1 : Vt0;
    int nloc = wave * 16;                         // 8 tiles / 8 waves per pass
    v8f acc = {};
    const __bf16* bRow = buf + (nloc + (lane & 15)) * VT_STRIDE;
    #pragma unroll
    for (int k0 = 0; k0 < 64; k0 += 32) {
      int off = k0 + sub;
      v16bf af = bf_combine(ld8(aRow + off), ld8(aRow + off + 16));
      v16bf bf = bf_combine(ld8(bRow + off), ld8(bRow + off + 16));
      acc = wmma_bf16(af, bf, acc);
    }
    int c  = p * VT_COLS + nloc + (lane & 15);
    int hb = (lane >> 4) ? 8 : 0;
    #pragma unroll
    for (int r = 0; r < 8; r++) {
      int h = hb + r;
      if (h < 8) VB[(long)b * 8192 + h * 1024 + c] = (__bf16)acc[r];
    }
    __syncthreads();                               // done reading buf
    if (p + 2 < 1024 / VT_COLS)
      stage_v(Vb, (__bf16*)buf, p + 2, tid, 256);  // refill for pass p+2
    __syncthreads();
  }
}

// ---------------------------------------------------------------------------
extern "C" void kernel_launch(void* const* d_in, const int* in_sizes, int n_in,
                              void* d_out, int out_size, void* d_ws, size_t ws_size,
                              hipStream_t stream)
{
  const float* q  = (const float*)d_in[0];
  const float* k  = (const float*)d_in[1];
  const float* v  = (const float*)d_in[2];
  const float* Wq = (const float*)d_in[3];
  const float* Wk = (const float*)d_in[4];
  const float* Wv = (const float*)d_in[5];
  const float* Wb = (const float*)d_in[6];
  const float* Wo = (const float*)d_in[7];
  const float* bo = (const float*)d_in[8];
  float* out = (float*)d_out;
  __bf16* ws = (__bf16*)d_ws;

  // workspace layout (bf16 elements); T and VBAR alias (T consumed per-batch
  // before VBAR for that batch is produced inside attn_kernel)
  __bf16* qbf = ws + 0L;
  __bf16* Wqb = ws + 2097152L;
  __bf16* Wbb = ws + 3145728L;
  __bf16* WkT = ws + 3162112L;
  __bf16* Wvb = ws + 4210688L;
  __bf16* Wob = ws + 5259264L;
  __bf16* QP  = ws + 6307840L;
  __bf16* QB  = ws + 8404992L;
  __bf16* TV  = ws + 10502144L;   // T[b][8][1024], later VBAR[b][8][1024]
  __bf16* X   = ws + 27279360L;

  prep_kernel<<<8192, 256, 0, stream>>>(q, Wq, Wk, Wv, Wb, Wo,
                                        qbf, Wqb, WkT, Wvb, Wbb, Wob);

  // QP = q @ Wq^T                          (M=2048, N=1024, K=1024)
  gemm16_kernel<<<dim3(16,128,1),128,0,stream>>>(qbf,1024,0,  Wqb,1024,0,
                                                 QP,nullptr,nullptr, 1024,0, 1024);
  // QB_h = QP_h @ Wb^T                     (per head: N=128, K=128)
  gemm16_kernel<<<dim3(2,128,8),128,0,stream>>>(QP,1024,128, Wbb,128,0,
                                                 QB,nullptr,nullptr, 1024,128, 128);
  // T_h = QB_h @ Wk_h                      (per head: N=1024, K=128; Bt = WkT)
  gemm16_kernel<<<dim3(16,128,8),128,0,stream>>>(QB,1024,128, WkT,1024,128,
                                                 TV,nullptr,nullptr, 8192,1024, 128);
  // fused: scores (vs raw k), softmax, vbar (vs raw v) — streams 1.07 GB once
  attn_kernel<<<2048, 256, 0, stream>>>(k, v, TV, TV);
  // X_h = VBAR_h @ Wv_h^T                  (per head: N=128, K=1024)
  gemm16_kernel<<<dim3(2,128,8),128,0,stream>>>(TV,8192,1024, Wvb,1024,131072,
                                                 X,nullptr,nullptr, 1024,128, 1024);
  // out = X @ Wo^T + bo                    (fp32 output)
  gemm16_kernel<<<dim3(16,128,1),128,0,stream>>>(X,1024,0,   Wob,1024,0,
                                                 nullptr,out,bo, 1024,0, 1024);
}
